// CausalSelfAttn_10831907520740
// MI455X (gfx1250) — matrix operand order repaired and checked by reading, not compiled, not run on hardware
//
#include <hip/hip_runtime.h>
#include <hip/hip_bf16.h>

// ---------------------------------------------------------------------------
// CDNA5 (gfx1250) causal self-attention, bf16 WMMA pipeline.
//   N=4096, D=2048, H=16, HD=128
//   Stage 0: convert x and W{q,k,v,o} f32 -> bf16 (one-shot, makes the GEMM
//            staging a pure byte move so GLOBAL_LOAD_ASYNC_TO_LDS applies)
//   Stage 1: Q = x@Wq^T+bq, K = x@Wk^T+bk   -> bf16 [H][N][HD]
//            V = x@Wv^T+bv                  -> bf16 [H][HD][N]  (transposed)
//   Stage 2: flash attention (online softmax), O -> bf16 [N][D]
//   Stage 3: out = O@Wo^T+bo -> f32 [N][D]
// GEMMs: v_wmma_f32_16x16x32_bf16 (wave32), async double-buffered LDS tiles.
// ---------------------------------------------------------------------------

typedef __attribute__((ext_vector_type(16))) __bf16 v16bf;
typedef __attribute__((ext_vector_type(8)))  __bf16 v8bf;
typedef __attribute__((ext_vector_type(8)))  float  v8f;

static constexpr int N_  = 4096;
static constexpr int D_  = 2048;
static constexpr int H_  = 16;
static constexpr int HD_ = 128;

#if defined(__has_builtin)
#if __has_builtin(__builtin_amdgcn_global_load_async_to_lds_b128) && \
    __has_builtin(__builtin_amdgcn_s_wait_asynccnt)
#define HAS_ASYNC 1
#endif
#endif
#ifndef HAS_ASYNC
#define HAS_ASYNC 0
#endif

__device__ __forceinline__ v8f vzero8() {
  v8f z;
#pragma unroll
  for (int i = 0; i < 8; ++i) z[i] = 0.0f;
  return z;
}

__device__ __forceinline__ v8f wmma_bf16(v16bf a, v16bf b, v8f c) {
  // D = A(16x32 bf16) * B(32x16 bf16) + C(16x16 f32)
  return __builtin_amdgcn_wmma_f32_16x16x32_bf16(
      /*neg_a=*/false, a, /*neg_b=*/false, b,
      /*c_mod=*/(short)0, c, /*reuse_a=*/false, /*reuse_b=*/false);
}

// A-matrix 16x32 (16-bit) per-lane K index for element i (ISA 7.12.2).
__device__ __forceinline__ int kidxA(int half, int i) {
  return ((i < 8) ? 0 : 16) + half * 8 + (i & 7);
}

__device__ __forceinline__ void ld8bf(const __bf16* __restrict__ p, __bf16* d) {
  const v8bf v = *(const v8bf*)p;
#pragma unroll
  for (int i = 0; i < 8; ++i) d[i] = v[i];
}

__device__ __forceinline__ float rmax16(float v) {
#pragma unroll
  for (int off = 1; off < 16; off <<= 1)
    v = fmaxf(v, __shfl_xor(v, off, 32));
  return v;
}
__device__ __forceinline__ float rsum16(float v) {
#pragma unroll
  for (int off = 1; off < 16; off <<= 1)
    v += __shfl_xor(v, off, 32);
  return v;
}

#if HAS_ASYNC
// Builtin signature (from hipcc diagnostic): v4i addrspace(1)* src,
// v4i addrspace(3)* dst, imm offset, imm cpol.
typedef int v4i __attribute__((vector_size(16)));
typedef __attribute__((address_space(1))) v4i* gv4i;
typedef __attribute__((address_space(3))) v4i* lv4i;

__device__ __forceinline__ void async_cp16(const __bf16* g, __bf16* l) {
  __builtin_amdgcn_global_load_async_to_lds_b128((gv4i)g, (lv4i)l, 0, 0);
}
// Copy 32 bytes (two disjoint 16B runs) global -> LDS, tracked by ASYNCcnt.
__device__ __forceinline__ void async_cp_2x16(const __bf16* g0,
                                              const __bf16* g1, v16bf* l) {
  __bf16* lp = (__bf16*)l;
  async_cp16(g0, lp);
  async_cp16(g1, lp + 8);
}
#endif

// ---------------------------------------------------------------------------
// f32 -> bf16 bulk convert (8 elems/thread/step)
// ---------------------------------------------------------------------------
__global__ __launch_bounds__(256) void cvt_bf16_kernel(
    const float* __restrict__ in, __bf16* __restrict__ out, int n) {
  const int stride = (int)(gridDim.x * blockDim.x) * 8;
  for (int i = (int)(blockIdx.x * blockDim.x + threadIdx.x) * 8; i < n;
       i += stride) {
    const float4 a = *(const float4*)(in + i);
    const float4 b = *(const float4*)(in + i + 4);
    v8bf o;
    o[0] = (__bf16)a.x; o[1] = (__bf16)a.y; o[2] = (__bf16)a.z; o[3] = (__bf16)a.w;
    o[4] = (__bf16)b.x; o[5] = (__bf16)b.y; o[6] = (__bf16)b.z; o[7] = (__bf16)b.w;
    *(v8bf*)(out + i) = o;
  }
}

// ---------------------------------------------------------------------------
// Projection GEMM: out = X @ W^T + bias     (X:[N,D] bf16, W:[D,D] bf16)
// 128x128 output tile / block (256 threads = 8 waves, each wave 32x64).
// Async double-buffered LDS tiles in WMMA fragment order.
// OMODE 0: bf16 out, head layout [H][N][HD]
// OMODE 1: bf16 out, transposed head layout [H][HD][N]   (for V)
// OMODE 2: f32  out, row major [N][D]                    (final output)
// ---------------------------------------------------------------------------
template <int OMODE>
__global__ __launch_bounds__(256) void proj_kernel(
    const __bf16* __restrict__ X, const __bf16* __restrict__ W,
    const float* __restrict__ bias, void* __restrict__ outp) {
  __shared__ v16bf ldsA[2][256];  // 8 row-subtiles x 32 lanes, fragment order
  __shared__ v16bf ldsB[2][256];  // 8 col-subtiles x 32 lanes, fragment order

  const int t    = threadIdx.x;
  const int lane = t & 31;
  const int wave = t >> 5;
  const int hf   = lane >> 4;
  const int nn   = lane & 15;
  const int wave_row = wave & 3;   // 4 row groups (32 rows each)
  const int wave_col = wave >> 2;  // 2 col groups (64 cols each)

  const int d0   = blockIdx.x * 128;
  const int row0 = blockIdx.y * 128;
  const int s    = wave;  // staging subtile handled by this thread's wave

  v8f acc[2][4];
#pragma unroll
  for (int rs = 0; rs < 2; ++rs)
#pragma unroll
    for (int cs = 0; cs < 4; ++cs) acc[rs][cs] = vzero8();

  const __bf16* arow = X + (size_t)(row0 + s * 16 + nn) * D_;
  const __bf16* brow = W + (size_t)(d0 + s * 16 + nn) * D_;

  auto compute = [&](int b) {
    v16bf af[2], bf_[4];
#pragma unroll
    for (int rs = 0; rs < 2; ++rs)
      af[rs] = ldsA[b][(wave_row * 2 + rs) * 32 + lane];
#pragma unroll
    for (int cs = 0; cs < 4; ++cs)
      bf_[cs] = ldsB[b][(wave_col * 4 + cs) * 32 + lane];
#pragma unroll
    for (int rs = 0; rs < 2; ++rs)
#pragma unroll
      for (int cs = 0; cs < 4; ++cs)
        acc[rs][cs] = wmma_bf16(af[rs], bf_[cs], acc[rs][cs]);
  };

#if HAS_ASYNC
  auto issue = [&](int k0, int b) {
    async_cp_2x16(arow + k0 + hf * 8, arow + k0 + 16 + hf * 8,
                  &ldsA[b][s * 32 + lane]);
    async_cp_2x16(brow + k0 + hf * 8, brow + k0 + 16 + hf * 8,
                  &ldsB[b][s * 32 + lane]);
  };
  issue(0, 0);
  int buf = 0;
  for (int k0 = 0; k0 < D_; k0 += 32) {
    if (k0 + 32 < D_) {
      issue(k0 + 32, buf ^ 1);
      __builtin_amdgcn_s_wait_asynccnt(4);  // current buffer's 4 ops done
    } else {
      __builtin_amdgcn_s_wait_asynccnt(0);
    }
    __syncthreads();
    compute(buf);
    __syncthreads();
    buf ^= 1;
  }
#else
  for (int k0 = 0; k0 < D_; k0 += 32) {
    {
      __bf16 tmp[16];
      ld8bf(arow + k0 + hf * 8, tmp);
      ld8bf(arow + k0 + 16 + hf * 8, tmp + 8);
      v16bf va;
#pragma unroll
      for (int i = 0; i < 16; ++i) va[i] = tmp[i];
      ldsA[0][s * 32 + lane] = va;
      ld8bf(brow + k0 + hf * 8, tmp);
      ld8bf(brow + k0 + 16 + hf * 8, tmp + 8);
      v16bf vb;
#pragma unroll
      for (int i = 0; i < 16; ++i) vb[i] = tmp[i];
      ldsB[0][s * 32 + lane] = vb;
    }
    __syncthreads();
    compute(0);
    __syncthreads();
  }
#endif

  // ---- epilogue: bias + store ----
#pragma unroll
  for (int rs = 0; rs < 2; ++rs) {
    const int rowbase = row0 + (wave_row * 2 + rs) * 16 + hf * 8;
#pragma unroll
    for (int cs = 0; cs < 4; ++cs) {
      const int dcol = d0 + (wave_col * 4 + cs) * 16 + nn;
      const float bv = bias[dcol];
#pragma unroll
      for (int r = 0; r < 8; ++r) {
        const float v = acc[rs][cs][r] + bv;
        const int row = rowbase + r;
        if (OMODE == 0) {
          ((__bf16*)outp)[((size_t)(dcol >> 7) * N_ + row) * HD_ + (dcol & 127)] =
              (__bf16)v;
        } else if (OMODE == 1) {
          ((__bf16*)outp)[((size_t)(dcol >> 7) * HD_ + (dcol & 127)) * N_ + row] =
              (__bf16)v;
        } else {
          ((float*)outp)[(size_t)row * D_ + dcol] = v;
        }
      }
    }
  }
}

// ---------------------------------------------------------------------------
// Flash attention: one block = 128 query rows of one head; wave w owns rows
// [q0+16w, q0+16w+16). Causal loop over 32-key chunks.
// ---------------------------------------------------------------------------
__global__ __launch_bounds__(256) void flash_kernel(
    const __bf16* __restrict__ Q,   // [H][N][HD]
    const __bf16* __restrict__ K,   // [H][N][HD]
    const __bf16* __restrict__ Vt,  // [H][HD][N]
    __bf16* __restrict__ O) {       // [N][D]
  __shared__ __bf16 ldsP[8][16 * 32];  // per-wave P re-layout scratch

  const int t    = threadIdx.x;
  const int lane = t & 31;
  const int wave = t >> 5;
  const int hf   = lane >> 4;
  const int nn   = lane & 15;
  const int h    = blockIdx.y;
  const int qrow0 = blockIdx.x * 128 + wave * 16;

  constexpr float SCALE = 0.08838834764831845f;  // 1/sqrt(128)

  // Q fragments for the full HD=128 K-depth (4 x 16x32 bf16 tiles)
  v16bf qa[4];
  {
    const __bf16* qp = Q + ((size_t)h * N_ + qrow0 + nn) * HD_;
#pragma unroll
    for (int d4 = 0; d4 < 4; ++d4) {
      __bf16 tmp[16];
      ld8bf(qp + d4 * 32 + hf * 16, tmp);
      ld8bf(qp + d4 * 32 + hf * 16 + 8, tmp + 8);
      v16bf a;
#pragma unroll
      for (int i = 0; i < 16; ++i) a[i] = tmp[i];
      qa[d4] = a;
    }
  }

  v8f acc[8];
#pragma unroll
  for (int c = 0; c < 8; ++c) acc[c] = vzero8();
  float mr[8], lr[8];
#pragma unroll
  for (int r = 0; r < 8; ++r) { mr[r] = -3.0e38f; lr[r] = 0.0f; }

  const int kmax = qrow0 + 15;  // causal limit for this wave (uniform)
  for (int kb = 0; kb <= kmax; kb += 32) {
    // prefetch next key chunk toward the caches (global_prefetch_b8)
    if (kb + 32 <= kmax) {
      __builtin_prefetch(K + ((size_t)h * N_ + kb + 32 + lane) * HD_, 0, 0);
      __builtin_prefetch(Vt + ((size_t)h * HD_ + lane * 4) * N_ + kb + 32, 0, 0);
    }

    // ---- S = Q K^T for 32 keys (two 16x16 f32 tiles) ----
    v8f s[2];
    s[0] = vzero8();
    s[1] = vzero8();
#pragma unroll
    for (int sub = 0; sub < 2; ++sub) {
#pragma unroll
      for (int d4 = 0; d4 < 4; ++d4) {
        const v16bf b = *(const v16bf*)(
            K + ((size_t)h * N_ + kb + sub * 16 + nn) * HD_ + d4 * 32 + hf * 16);
        s[sub] = wmma_bf16(qa[d4], b, s[sub]);
      }
    }

    // ---- scale + causal mask (mask input is deterministic; not read) ----
    float p[2][8];
#pragma unroll
    for (int sub = 0; sub < 2; ++sub) {
      const int kj = kb + sub * 16 + nn;
#pragma unroll
      for (int r = 0; r < 8; ++r) {
        const int qi = qrow0 + hf * 8 + r;
        const float v = s[sub][r] * SCALE;
        p[sub][r] = (kj > qi) ? -3.0e38f : v;
      }
    }

    // ---- online softmax update ----
    float alpha[8], mnew[8];
#pragma unroll
    for (int r = 0; r < 8; ++r) {
      const float rm = rmax16(fmaxf(p[0][r], p[1][r]));
      mnew[r] = fmaxf(mr[r], rm);
      alpha[r] = __expf(mr[r] - mnew[r]);
      mr[r] = mnew[r];
    }
#pragma unroll
    for (int sub = 0; sub < 2; ++sub)
#pragma unroll
      for (int r = 0; r < 8; ++r)
        p[sub][r] = __expf(p[sub][r] - mnew[r]);
#pragma unroll
    for (int r = 0; r < 8; ++r)
      lr[r] = lr[r] * alpha[r] + rsum16(p[0][r] + p[1][r]);
#pragma unroll
    for (int c = 0; c < 8; ++c)
#pragma unroll
      for (int r = 0; r < 8; ++r) acc[c][r] *= alpha[r];

    // ---- P: C-layout -> A-layout via per-wave LDS ----
    __bf16* myP = &ldsP[wave][0];
#pragma unroll
    for (int sub = 0; sub < 2; ++sub)
#pragma unroll
      for (int r = 0; r < 8; ++r)
        myP[(hf * 8 + r) * 32 + sub * 16 + nn] = (__bf16)p[sub][r];
    v16bf pa;
#pragma unroll
    for (int i = 0; i < 16; ++i) pa[i] = myP[nn * 32 + hf * 16 + i];

    // ---- O += P @ V (Vt gives contiguous 32B B-fragments) ----
#pragma unroll
    for (int c = 0; c < 8; ++c) {
      const v16bf b = *(const v16bf*)(
          Vt + ((size_t)h * HD_ + c * 16 + nn) * N_ + kb + hf * 16);
      acc[c] = wmma_bf16(pa, b, acc[c]);
    }
  }

  // ---- normalize + store O ----
#pragma unroll
  for (int r = 0; r < 8; ++r) {
    const float inv = 1.0f / lr[r];
    const size_t row = (size_t)(qrow0 + hf * 8 + r);
#pragma unroll
    for (int c = 0; c < 8; ++c)
      O[row * D_ + h * HD_ + c * 16 + nn] = (__bf16)(acc[c][r] * inv);
  }
}

// ---------------------------------------------------------------------------
extern "C" void kernel_launch(void* const* d_in, const int* in_sizes, int n_in,
                              void* d_out, int out_size, void* d_ws,
                              size_t ws_size, hipStream_t stream) {
  (void)in_sizes; (void)n_in; (void)out_size; (void)ws_size;

  const float* x  = (const float*)d_in[0];
  // d_in[1] = attn_mask: deterministic causal mask, computed analytically.
  const float* Wq = (const float*)d_in[2];
  const float* bq = (const float*)d_in[3];
  const float* Wk = (const float*)d_in[4];
  const float* bk = (const float*)d_in[5];
  const float* Wv = (const float*)d_in[6];
  const float* bv = (const float*)d_in[7];
  const float* Wo = (const float*)d_in[8];
  const float* bo = (const float*)d_in[9];

  char* ws = (char*)d_ws;
  const size_t MB = 1024 * 1024;
  __bf16* qb  = (__bf16*)(ws + 0 * MB);    // [H][N][HD]   16MB
  __bf16* kb  = (__bf16*)(ws + 16 * MB);   // [H][N][HD]   16MB
  __bf16* vtb = (__bf16*)(ws + 32 * MB);   // [H][HD][N]   16MB
  __bf16* ob  = (__bf16*)(ws + 48 * MB);   // [N][D]       16MB
  __bf16* xb  = (__bf16*)(ws + 64 * MB);   // [N][D]       16MB
  __bf16* wqb = (__bf16*)(ws + 80 * MB);   // [D][D]        8MB
  __bf16* wkb = (__bf16*)(ws + 88 * MB);
  __bf16* wvb = (__bf16*)(ws + 96 * MB);
  __bf16* wob = (__bf16*)(ws + 104 * MB);

  const dim3 blk(256);
  const int nx = N_ * D_, nw = D_ * D_;
  cvt_bf16_kernel<<<dim3(nx / (8 * 256)), blk, 0, stream>>>(x, xb, nx);
  cvt_bf16_kernel<<<dim3(nw / (8 * 256)), blk, 0, stream>>>(Wq, wqb, nw);
  cvt_bf16_kernel<<<dim3(nw / (8 * 256)), blk, 0, stream>>>(Wk, wkb, nw);
  cvt_bf16_kernel<<<dim3(nw / (8 * 256)), blk, 0, stream>>>(Wv, wvb, nw);
  cvt_bf16_kernel<<<dim3(nw / (8 * 256)), blk, 0, stream>>>(Wo, wob, nw);

  const dim3 gproj(D_ / 128, N_ / 128);
  const dim3 gattn(N_ / 128, H_);
  proj_kernel<0><<<gproj, blk, 0, stream>>>(xb, wqb, bq, (void*)qb);
  proj_kernel<0><<<gproj, blk, 0, stream>>>(xb, wkb, bk, (void*)kb);
  proj_kernel<1><<<gproj, blk, 0, stream>>>(xb, wvb, bv, (void*)vtb);
  flash_kernel<<<gattn, blk, 0, stream>>>(qb, kb, vtb, ob);
  proj_kernel<2><<<gproj, blk, 0, stream>>>(ob, wob, bo, d_out);
}
